// MultiHeadAttention_60413009985887
// MI455X (gfx1250) — compile-verified, matrix operand order checked by reading, not run on hardware
//
#include <hip/hip_runtime.h>

#define B_  4
#define S_  2048
#define E_  1024
#define H_  16
#define DK_ 64

typedef unsigned short u16;
typedef __attribute__((ext_vector_type(16))) __bf16 v16bf;
typedef __attribute__((ext_vector_type(8)))  float  v8f;

union Frag16 { v16bf v; uint4 q[2]; };

static __device__ __forceinline__ u16 f2bf(float f) {
    union { float f; unsigned u; } x; x.f = f;
    unsigned r = x.u + 0x7FFFu + ((x.u >> 16) & 1u);   // round-to-nearest-even
    return (u16)(r >> 16);
}

static __device__ __forceinline__ v8f wmma_bf16(v16bf a, v16bf b, v8f c) {
    // v_wmma_f32_16x16x32_bf16: D = A(16x32) * B(32x16) + C(16x16 f32)
    return __builtin_amdgcn_wmma_f32_16x16x32_bf16(
        false, a, false, b, (short)0, c, false, false);
}

// A/B fragment per lane: row = lane&15; element j -> K = (lane>>4)*8 + j + (j>=8 ? 8 : 0)
// i.e. two contiguous 8-element (16B) chunks at K-offsets hi*8 and hi*8+16.
static __device__ __forceinline__ void load_frag(Frag16& f, const u16* p) {
    f.q[0] = *(const uint4*)(p);
    f.q[1] = *(const uint4*)(p + 16);
}

// ---------------------------------------------------------------- convert f32 -> bf16
__global__ void __launch_bounds__(256) cvt_f32_bf16(const float* __restrict__ in,
                                                    u16* __restrict__ out, size_t n) {
    size_t i = (size_t)blockIdx.x * blockDim.x + threadIdx.x;
    if (i < n) out[i] = f2bf(in[i]);
}

// ---------------------------------------------------------------- GEMM: C = A[M,K] * Bw[N,K]^T
// mode 0: bf16 row-major C[M,N]
// mode 1: f32  row-major C[M,N]
// mode 2: bf16 "V-transpose" store: (m=b*S+s, n=h*64+d) -> VT[((b*H+h)*DK+d)*S + s]
__global__ void __launch_bounds__(256) gemm_bf16_wmma(const u16* __restrict__ A,
                                                      const u16* __restrict__ Bw,
                                                      void* __restrict__ Cout,
                                                      int M, int N, int K, int mode) {
    const int lane = threadIdx.x & 31;
    const int wave = threadIdx.x >> 5;
    const int lrow = lane & 15;
    const int hi   = lane >> 4;
    const int mBase = blockIdx.y * 128 + (wave >> 2) * 64;   // 2 waves along M
    const int nBase = blockIdx.x * 128 + (wave & 3)  * 32;   // 4 waves along N

    v8f acc[4][2];
#pragma unroll
    for (int t = 0; t < 4; ++t)
#pragma unroll
        for (int u = 0; u < 2; ++u) { v8f z = {}; acc[t][u] = z; }

    for (int k0 = 0; k0 < K; k0 += 32) {
        Frag16 af[4], bf[2];
#pragma unroll
        for (int t = 0; t < 4; ++t)
            load_frag(af[t], A + (size_t)(mBase + t * 16 + lrow) * K + k0 + hi * 8);
#pragma unroll
        for (int u = 0; u < 2; ++u)
            load_frag(bf[u], Bw + (size_t)(nBase + u * 16 + lrow) * K + k0 + hi * 8);
#pragma unroll
        for (int t = 0; t < 4; ++t)
#pragma unroll
            for (int u = 0; u < 2; ++u)
                acc[t][u] = wmma_bf16(af[t].v, bf[u].v, acc[t][u]);
    }

    // C/D layout: lane l -> n = nTile + (l&15); VGPR r -> m = mTile + r + 8*(l>>4)
#pragma unroll
    for (int t = 0; t < 4; ++t)
#pragma unroll
        for (int u = 0; u < 2; ++u)
#pragma unroll
            for (int r = 0; r < 8; ++r) {
                const int m = mBase + t * 16 + r + 8 * hi;
                const int n = nBase + u * 16 + lrow;
                const float val = acc[t][u][r];
                if (mode == 1) {
                    ((float*)Cout)[(size_t)m * N + n] = val;
                } else if (mode == 0) {
                    ((u16*)Cout)[(size_t)m * N + n] = f2bf(val);
                } else {
                    const int b = m >> 11, s = m & (S_ - 1);
                    const int h = n >> 6,  d = n & (DK_ - 1);
                    ((u16*)Cout)[(((size_t)b * H_ + h) * DK_ + d) * S_ + s] = f2bf(val);
                }
            }
}

// ---------------------------------------------------------------- flash attention
// One wave handles 16 query rows for one (b,h). Streams keys 32 at a time.
__global__ void __launch_bounds__(256) attn_wmma(const u16* __restrict__ Q,
                                                 const u16* __restrict__ Km,
                                                 const u16* __restrict__ VT,
                                                 u16* __restrict__ ctx) {
    __shared__ u16 sP[8][16 * 32];   // per-wave P scratch (D-layout -> A-layout transpose)

    const int lane = threadIdx.x & 31;
    const int wave = threadIdx.x >> 5;
    const int lrow = lane & 15;
    const int hi   = lane >> 4;
    const int gw   = blockIdx.x * 8 + wave;   // global wave id
    const int bh   = gw >> 7;                 // 128 q-tiles per (b,h)
    const int qt   = gw & 127;
    const int b    = bh >> 4;
    const int h    = bh & 15;
    const int q0   = qt << 4;
    const float SC = 0.125f;                  // 1/sqrt(DK)

    Frag16 aq[2];
#pragma unroll
    for (int kd = 0; kd < 2; ++kd)
        load_frag(aq[kd], Q + ((size_t)b * S_ + q0 + lrow) * E_ + h * DK_ + kd * 32 + hi * 8);

    v8f acc[4];
#pragma unroll
    for (int t = 0; t < 4; ++t) { v8f z = {}; acc[t] = z; }
    float vmax[8], vsum[8];
#pragma unroll
    for (int r = 0; r < 8; ++r) { vmax[r] = -__builtin_inff(); vsum[r] = 0.0f; }

    for (int k0 = 0; k0 < S_; k0 += 32) {
        Frag16 bk[2][2];
#pragma unroll
        for (int u = 0; u < 2; ++u)
#pragma unroll
            for (int kd = 0; kd < 2; ++kd)
                load_frag(bk[u][kd],
                          Km + ((size_t)b * S_ + k0 + u * 16 + lrow) * E_ + h * DK_ + kd * 32 + hi * 8);

        v8f s0 = {}, s1 = {};
        s0 = wmma_bf16(aq[0].v, bk[0][0].v, s0);
        s0 = wmma_bf16(aq[1].v, bk[0][1].v, s0);
        s1 = wmma_bf16(aq[0].v, bk[1][0].v, s1);
        s1 = wmma_bf16(aq[1].v, bk[1][1].v, s1);

        float corr[8];
#pragma unroll
        for (int r = 0; r < 8; ++r) {
            float a0 = s0[r] * SC, a1 = s1[r] * SC;
            float v = fmaxf(a0, a1);
            v = fmaxf(v, __shfl_xor(v, 1, 32));
            v = fmaxf(v, __shfl_xor(v, 2, 32));
            v = fmaxf(v, __shfl_xor(v, 4, 32));
            v = fmaxf(v, __shfl_xor(v, 8, 32));      // row-max within 16-lane half
            const float mo = vmax[r];
            const float mn = fmaxf(mo, v);
            const float c  = __expf(mo - mn);        // exp(-inf)=0 on first tile
            const float p0 = __expf(a0 - mn);
            const float p1 = __expf(a1 - mn);
            float rs = p0 + p1;
            rs += __shfl_xor(rs, 1, 32);
            rs += __shfl_xor(rs, 2, 32);
            rs += __shfl_xor(rs, 4, 32);
            rs += __shfl_xor(rs, 8, 32);
            vsum[r] = vsum[r] * c + rs;
            vmax[r] = mn;
            corr[r] = c;
            const int m = r + 8 * hi;                // D-layout row
            sP[wave][m * 32 + lrow]      = f2bf(p0);
            sP[wave][m * 32 + 16 + lrow] = f2bf(p1);
        }
#pragma unroll
        for (int t = 0; t < 4; ++t)
#pragma unroll
            for (int r = 0; r < 8; ++r) acc[t][r] *= corr[r];

        Frag16 pf;                                   // reload P in A-fragment layout
        pf.q[0] = *(const uint4*)&sP[wave][lrow * 32 + hi * 8];
        pf.q[1] = *(const uint4*)&sP[wave][lrow * 32 + hi * 8 + 16];

#pragma unroll
        for (int t = 0; t < 4; ++t) {
            Frag16 vf;   // B-operand: lane = d, contiguous keys thanks to [B,H,DK,S] layout
            load_frag(vf, VT + (((size_t)b * H_ + h) * DK_ + t * 16 + lrow) * S_ + k0 + hi * 8);
            acc[t] = wmma_bf16(pf.v, vf.v, acc[t]);
        }
    }

#pragma unroll
    for (int r = 0; r < 8; ++r) {
        const float inv = 1.0f / vsum[r];
        const int m = q0 + r + 8 * hi;
#pragma unroll
        for (int t = 0; t < 4; ++t)
            ctx[((size_t)b * S_ + m) * E_ + h * DK_ + t * 16 + lrow] = f2bf(acc[t][r] * inv);
    }
}

// ---------------------------------------------------------------- launch
extern "C" void kernel_launch(void* const* d_in, const int* in_sizes, int n_in,
                              void* d_out, int out_size, void* d_ws, size_t ws_size,
                              hipStream_t stream) {
    (void)in_sizes; (void)n_in; (void)out_size; (void)ws_size;
    const float* x  = (const float*)d_in[0];
    const float* Wq = (const float*)d_in[1];
    const float* Wk = (const float*)d_in[2];
    const float* Wv = (const float*)d_in[3];
    const float* Wo = (const float*)d_in[4];

    const size_t ME = (size_t)B_ * S_;        // 8192 rows
    const size_t EE = (size_t)E_ * E_;

    u16* xb  = (u16*)d_ws;                    // [ME, E]     bf16 x
    u16* wqb = xb  + ME * E_;                 // [E, E] each
    u16* wkb = wqb + EE;
    u16* wvb = wkb + EE;
    u16* wob = wvb + EE;
    u16* Qb  = wob + EE;                      // [B,S,E]
    u16* Kb  = Qb  + ME * E_;                 // [B,S,E]
    u16* VT  = Kb  + ME * E_;                 // [B,H,DK,S]
    u16* Cx  = VT  + ME * E_;                 // [B,S,E] attention output

    cvt_f32_bf16<<<(unsigned)((ME * E_ + 255) / 256), 256, 0, stream>>>(x,  xb,  ME * E_);
    cvt_f32_bf16<<<(unsigned)((EE + 255) / 256),      256, 0, stream>>>(Wq, wqb, EE);
    cvt_f32_bf16<<<(unsigned)((EE + 255) / 256),      256, 0, stream>>>(Wk, wkb, EE);
    cvt_f32_bf16<<<(unsigned)((EE + 255) / 256),      256, 0, stream>>>(Wv, wvb, EE);
    cvt_f32_bf16<<<(unsigned)((EE + 255) / 256),      256, 0, stream>>>(Wo, wob, EE);

    dim3 blk(256);
    dim3 grd(E_ / 128, (unsigned)(ME / 128));   // (8, 64)
    gemm_bf16_wmma<<<grd, blk, 0, stream>>>(xb, wqb, Qb,          (int)ME, E_, E_, 0);
    gemm_bf16_wmma<<<grd, blk, 0, stream>>>(xb, wkb, Kb,          (int)ME, E_, E_, 0);
    gemm_bf16_wmma<<<grd, blk, 0, stream>>>(xb, wvb, VT,          (int)ME, E_, E_, 2);

    attn_wmma<<<(B_ * H_ * (S_ / 16)) / 8, 256, 0, stream>>>(Qb, Kb, VT, Cx);

    gemm_bf16_wmma<<<grd, blk, 0, stream>>>(Cx, wob, d_out,       (int)ME, E_, E_, 1);
}
